// GRUGenerator_6846177870161
// MI455X (gfx1250) — matrix operand order repaired and checked by reading, not compiled, not run on hardware
//
#include <hip/hip_runtime.h>
#include <hip/hip_bf16.h>

#define BROWS 8192
#define INSZ  256
#define H     128
#define STEPS 50

typedef __attribute__((ext_vector_type(16))) _Float16 v16h;
typedef __attribute__((ext_vector_type(8)))  _Float16 v8h;
typedef __attribute__((ext_vector_type(8)))  float    v8f;

static __device__ __forceinline__ v16h cat8(v8h lo, v8h hi) {
  return __builtin_shufflevector(lo, hi, 0,1,2,3,4,5,6,7,8,9,10,11,12,13,14,15);
}

// A-fragment 16x32 f16 from row-major [16 x S] tile at column kbase.
// Lanes 0-15: M=lane, K = kb+0..7 and kb+16..23 ; lanes 16-31: M=lane-16, K = kb+8..15 and kb+24..31.
template<int S>
static __device__ __forceinline__ v16h load_a(const _Float16* p, int lane, int kbase) {
  int row = lane & 15;
  int k0  = kbase + ((lane < 16) ? 0 : 8);
  v8h lo = *(const v8h*)(p + row * S + k0);
  v8h hi = *(const v8h*)(p + row * S + k0 + 16);
  return cat8(lo, hi);
}

// B-fragment 32x16 f16: element (k,n) stored at p[n*S + k] (k-contiguous rows).
// Lanes 0-15: N=lane, K = kb+0..7 and kb+16..23 ; lanes 16-31: N=lane-16, K = kb+8..15 and kb+24..31.
template<int S>
static __device__ __forceinline__ v16h load_b(const _Float16* p, int lane, int nbase, int kbase) {
  int n = nbase + (lane & 15);
  int k = kbase + ((lane < 16) ? 0 : 8);
  v8h lo = *(const v8h*)(p + n * S + k);
  v8h hi = *(const v8h*)(p + n * S + k + 16);
  return cat8(lo, hi);
}

static __device__ __forceinline__ v8f wmma(v16h a, v16h b, v8f c) {
  return __builtin_amdgcn_wmma_f32_16x16x32_f16(false, a, false, b, (short)0, c, false, false);
}

static __device__ __forceinline__ float sigmoidf(float x) {
  return 1.0f / (1.0f + __expf(-x));
}
static __device__ __forceinline__ float tanh_fast(float x) {
  x = fminf(fmaxf(x, -15.0f), 15.0f);
  float e = __expf(2.0f * x);
  return (e - 1.0f) / (e + 1.0f);
}

// One MLP layer for a 16-row tile: dst[16 x H] = act(A[16 x (NK*32)] * W + bias)
template<int NK, int SA, int SB>
static __device__ __forceinline__ void mlp_layer(const _Float16* aSrc, const _Float16* wT,
                                                 const float* bias, bool relu,
                                                 _Float16* dst, int lane) {
  v16h a[NK];
#pragma unroll
  for (int kc = 0; kc < NK; ++kc) a[kc] = load_a<SA>(aSrc, lane, kc * 32);
#pragma unroll
  for (int n = 0; n < 8; ++n) {
    v8f acc = {};
#pragma unroll
    for (int kc = 0; kc < NK; ++kc)
      acc = wmma(a[kc], load_b<SB>(wT, lane, n * 16, kc * 32), acc);
    int col = n * 16 + (lane & 15);
    float bv = bias[col];
#pragma unroll
    for (int v = 0; v < 8; ++v) {
      int m = v + ((lane < 16) ? 0 : 8);
      float val = acc[v] + bv;
      if (relu) val = fmaxf(val, 0.0f);
      dst[m * H + col] = (_Float16)val;
    }
  }
}

// ---------------- prep: f32 -> f16 casts and weight transposes ----------------
__global__ void prep_kernel(const float* __restrict__ x,
                            const float* __restrict__ ew0, const float* __restrict__ ew1,
                            const float* __restrict__ ew2, const float* __restrict__ dw0,
                            const float* __restrict__ w_hh,
                            _Float16* __restrict__ xh,
                            _Float16* __restrict__ e0T, _Float16* __restrict__ e1T,
                            _Float16* __restrict__ e2T, _Float16* __restrict__ d0T,
                            _Float16* __restrict__ whh_h) {
  int i = blockIdx.x * blockDim.x + threadIdx.x;
  int stride = gridDim.x * blockDim.x;
  for (int idx = i; idx < BROWS * INSZ; idx += stride) xh[idx] = (_Float16)x[idx];
  if (i < H * INSZ) { int n = i >> 8, k = i & 255; e0T[i] = (_Float16)ew0[k * H + n]; }
  if (i < H * H)    { int n = i >> 7, k = i & 127; e1T[i] = (_Float16)ew1[k * H + n];
                                                   e2T[i] = (_Float16)ew2[k * H + n]; }
  if (i < 16 * H)   { int n = i >> 7, k = i & 127; d0T[i] = (_Float16)dw0[k * 16 + n]; }
  if (i < 3 * H * H) whh_h[i] = (_Float16)w_hh[i];   // already [384][128] k-contiguous
}

// ---------------- persistent fused encoder + GRU + decoder ----------------
// LDS layout (bytes): [0,98304) w_hh f16 [384][128]; [98304,131072) 8 h-tiles 16x128 f16;
// [131072,134656) gate param arrays (7 x 128 f32).
__global__ __launch_bounds__(256) void gru_main(
    const _Float16* __restrict__ xh,
    const _Float16* __restrict__ e0T, const _Float16* __restrict__ e1T,
    const _Float16* __restrict__ e2T, const _Float16* __restrict__ d0T,
    const _Float16* __restrict__ whh_h,
    const float* __restrict__ eb0, const float* __restrict__ eb1, const float* __restrict__ eb2,
    const float* __restrict__ w_ih,
    const float* __restrict__ b_ih, const float* __restrict__ b_hh,
    const float* __restrict__ db0, const float* __restrict__ dw1, const float* __restrict__ db1,
    float* __restrict__ out) {
  extern __shared__ char smem[];
  _Float16* whh_lds = (_Float16*)smem;                       // 49152 halfs @ LDS byte 0
  _Float16* htiles  = (_Float16*)smem + 3 * H * H;           // 8 * 2048 halfs
  float*    pw      = (float*)(smem + 131072);
  float* wrA  = pw;        float* wzA  = pw + 128; float* wnA  = pw + 256;
  float* brA  = pw + 384;  float* bzA  = pw + 512;
  float* binA = pw + 640;  float* bhnA = pw + 768;

  // ---- async fill of w_hh into LDS: 96KB, 16B/lane per issue, ASYNCcnt-tracked ----
  {
    unsigned base_off = threadIdx.x * 16u;   // byte offset, same in global and LDS
#pragma unroll 1
    for (int it = 0; it < 24; ++it) {        // 24 * 256 threads * 16B = 98304 B
      unsigned off = base_off + (unsigned)it * 4096u;
      asm volatile("global_load_async_to_lds_b128 %0, %1, %2"
                   :
                   : "v"(off), "v"(off), "s"(whh_h)
                   : "memory");
    }
  }
  if (threadIdx.x < H) {
    int c = threadIdx.x;
    wrA[c]  = w_ih[c];           wzA[c] = w_ih[H + c];       wnA[c] = w_ih[2 * H + c];
    brA[c]  = b_ih[c] + b_hh[c];
    bzA[c]  = b_ih[H + c] + b_hh[H + c];
    binA[c] = b_ih[2 * H + c];   bhnA[c] = b_hh[2 * H + c];
  }
  asm volatile("s_wait_asynccnt 0" ::: "memory");
  __syncthreads();

  int lane = threadIdx.x & 31;
  int wave = threadIdx.x >> 5;
  int tile = blockIdx.x * 8 + wave;     // 64 blocks * 8 waves = 512 tiles of 16 rows
  int row_base = tile * 16;
  _Float16* my_h = htiles + wave * 16 * H;

  // zero first output column for this tile's rows
  if (lane < 16) out[(size_t)(row_base + lane) * STEPS] = 0.0f;

  // ---- encoder MLP: x[16,256] -> h[16,128] in my_h ----
  mlp_layer<8, INSZ, INSZ>(xh + (size_t)row_base * INSZ, e0T, eb0, true,  my_h, lane);
  mlp_layer<4, H,    H   >(my_h,                         e1T, eb1, true,  my_h, lane);
  mlp_layer<4, H,    H   >(my_h,                         e2T, eb2, false, my_h, lane);

  // hoist decoder dw0^T B-fragments (constant over steps)
  v16h bd[4];
#pragma unroll
  for (int kc = 0; kc < 4; ++kc) bd[kc] = load_b<H>(d0T, lane, 0, kc * 32);
  float db0v = db0[lane & 15];
  float dw1v = dw1[lane & 15];
  float db1v = db1[0];

  float prevv[8];
#pragma unroll
  for (int v = 0; v < 8; ++v) prevv[v] = 0.0f;

  // A-fragments of h (held in VGPRs across the step's 24 gate tiles)
  v16h ah[4];
#pragma unroll
  for (int kc = 0; kc < 4; ++kc) ah[kc] = load_a<H>(my_h, lane, kc * 32);

#pragma unroll 1
  for (int t = 0; t < STEPS; ++t) {
    // ---- GRU gates: gh = h @ w_hh^T, 8 column-blocks x {r,z,n} x K=128 ----
#pragma unroll
    for (int j = 0; j < 8; ++j) {
      v8f accR = {}, accZ = {}, accN = {};
#pragma unroll
      for (int kc = 0; kc < 4; ++kc) {
        v16h brf = load_b<H>(whh_lds, lane,         j * 16, kc * 32);
        v16h bzf = load_b<H>(whh_lds, lane, H     + j * 16, kc * 32);
        v16h bnf = load_b<H>(whh_lds, lane, 2 * H + j * 16, kc * 32);
        accR = wmma(ah[kc], brf, accR);
        accZ = wmma(ah[kc], bzf, accZ);
        accN = wmma(ah[kc], bnf, accN);
      }
      int col = j * 16 + (lane & 15);
      float wrv = wrA[col], wzv = wzA[col], wnv = wnA[col];
      float brv = brA[col], bzv = bzA[col], binv = binA[col], bhnv = bhnA[col];
#pragma unroll
      for (int v = 0; v < 8; ++v) {
        int m = v + ((lane < 16) ? 0 : 8);
        float pm = prevv[v];
        float r  = sigmoidf(accR[v] + pm * wrv + brv);
        float z  = sigmoidf(accZ[v] + pm * wzv + bzv);
        float nn = tanh_fast(pm * wnv + binv + r * (accN[v] + bhnv));
        float hold = (float)my_h[m * H + col];
        float hnew = (1.0f - z) * nn + z * hold;
        my_h[m * H + col] = (_Float16)hnew;
      }
    }
    // reload A-fragments of new h (same-wave LDS ops are in-order)
#pragma unroll
    for (int kc = 0; kc < 4; ++kc) ah[kc] = load_a<H>(my_h, lane, kc * 32);

    // ---- decoder: d = relu(h @ dw0 + db0) [16x16]; out = d @ dw1 + db1 ----
    v8f d = {};
#pragma unroll
    for (int kc = 0; kc < 4; ++kc) d = wmma(ah[kc], bd[kc], d);
#pragma unroll
    for (int v = 0; v < 8; ++v) {
      float p = fmaxf(d[v] + db0v, 0.0f) * dw1v;
      p += __shfl_xor(p, 1);   // masks < 16: stay within 16-lane half
      p += __shfl_xor(p, 2);
      p += __shfl_xor(p, 4);
      p += __shfl_xor(p, 8);
      float outv = p + db1v;
      prevv[v] = outv;
      if ((lane & 15) == 0 && t < STEPS - 1) {
        int row = row_base + v + ((lane < 16) ? 0 : 8);
        out[(size_t)row * STEPS + t + 1] = outv;
      }
    }
  }
}

extern "C" void kernel_launch(void* const* d_in, const int* in_sizes, int n_in,
                              void* d_out, int out_size, void* d_ws, size_t ws_size,
                              hipStream_t stream) {
  (void)in_sizes; (void)n_in; (void)out_size; (void)ws_size;
  const float* x    = (const float*)d_in[0];
  const float* ew0  = (const float*)d_in[1];
  const float* eb0  = (const float*)d_in[2];
  const float* ew1  = (const float*)d_in[3];
  const float* eb1  = (const float*)d_in[4];
  const float* ew2  = (const float*)d_in[5];
  const float* eb2  = (const float*)d_in[6];
  const float* w_ih = (const float*)d_in[7];
  const float* w_hh = (const float*)d_in[8];
  const float* b_ih = (const float*)d_in[9];
  const float* b_hh = (const float*)d_in[10];
  const float* dw0  = (const float*)d_in[11];
  const float* db0  = (const float*)d_in[12];
  const float* dw1  = (const float*)d_in[13];
  const float* db1  = (const float*)d_in[14];

  char* ws = (char*)d_ws;
  _Float16* xh    = (_Float16*)(ws);               // 8192*256*2 = 4194304
  _Float16* e0T   = (_Float16*)(ws + 4194304);     // 128*256*2  = 65536
  _Float16* e1T   = (_Float16*)(ws + 4259840);     // 128*128*2  = 32768
  _Float16* e2T   = (_Float16*)(ws + 4292608);     // 128*128*2  = 32768
  _Float16* d0T   = (_Float16*)(ws + 4325376);     // 16*128*2   = 4096
  _Float16* whh_h = (_Float16*)(ws + 4329472);     // 384*128*2  = 98304

  prep_kernel<<<512, 256, 0, stream>>>(x, ew0, ew1, ew2, dw0, w_hh,
                                       xh, e0T, e1T, e2T, d0T, whh_h);

  const size_t smem = 134656;  // 96KB w_hh + 32KB h-tiles + 3.5KB gate params
  (void)hipFuncSetAttribute(reinterpret_cast<const void*>(&gru_main),
                            hipFuncAttributeMaxDynamicSharedMemorySize, (int)smem);
  gru_main<<<BROWS / (16 * 8), 256, smem, stream>>>(
      xh, e0T, e1T, e2T, d0T, whh_h, eb0, eb1, eb2,
      w_ih, b_ih, b_hh, db0, dw1, db1, (float*)d_out);
}
